// BoxEstimatorPointNetPlusPlusCuda_80599356277054
// MI455X (gfx1250) — matrix operand order repaired and checked by reading, not compiled, not run on hardware
//
#include <hip/hip_runtime.h>

typedef __attribute__((ext_vector_type(16))) _Float16 v16h;
typedef __attribute__((ext_vector_type(8)))  float    v8f;
typedef int v4i __attribute__((vector_size(16)));

#define BB   128
#define NN   2048
#define S1   128
#define NS1  64
#define S2   32
#define NS2  64

#if __has_builtin(__builtin_amdgcn_global_load_async_to_lds_b128) && \
    __has_builtin(__builtin_amdgcn_s_wait_asynccnt)
#define USE_ASYNC_LDS 1
#define GAS1(p) ((__attribute__((address_space(1))) v4i*)(p))
#define LAS3(p) ((__attribute__((address_space(3))) v4i*)(p))
#endif

// ---------------------------------------------------------------------------
// Fold eval-mode BN into conv/fc weights: Wf = W*sc (f16), bias = b*sc + shift
// ---------------------------------------------------------------------------
__global__ void fold_bn_kernel(const float* __restrict__ W, const float* __restrict__ bc,
                               const float* __restrict__ g, const float* __restrict__ bb,
                               const float* __restrict__ m, const float* __restrict__ v,
                               _Float16* __restrict__ Wf, float* __restrict__ biasf,
                               int M, int K, int Mpad, int Kpad)
{
    int tid = blockIdx.x * blockDim.x + threadIdx.x;
    int total = Mpad * Kpad;
    if (tid >= total) return;
    int mm = tid / Kpad, kk = tid % Kpad;
    float sc = 1.f, shift = 0.f;
    if (mm < M && g != nullptr) {
        float iv = rsqrtf(v[mm] + 1e-5f);
        sc = g[mm] * iv;
        shift = bb[mm] - m[mm] * sc;
    }
    float wv = (mm < M && kk < K) ? W[(size_t)mm * K + kk] * sc : 0.f;
    Wf[(size_t)mm * Kpad + kk] = (_Float16)wv;
    if (kk == 0) biasf[mm] = (mm < M) ? (bc[mm] * sc + shift) : 0.f;
}

// ---------------------------------------------------------------------------
// WMMA GEMM: Y[p, m] = act( sum_k Wf[m,k] * X[p,k] + bias[m] )
// Block = 256 threads = 8 wave32. Block tile: 128 points x 32 Cout.
// Each wave: 16 points x 32 Cout = two 16x16 WMMA accumulators sharing B.
// LDS-staged X tile (128x32 f16) + weight tile (32x32 f16), via
// GLOBAL_LOAD_ASYNC_TO_LDS_B128 (ASYNCcnt) when the builtin is available.
// ---------------------------------------------------------------------------
__global__ void __launch_bounds__(256)
gemm_wmma(const _Float16* __restrict__ Wf, const float* __restrict__ bias,
          const _Float16* __restrict__ X, void* __restrict__ Yv,
          int P, int Kpad, int Mvalid, int Mstride, int relu, int out_f32)
{
    __shared__ __align__(16) _Float16 xs[128 * 32];
    __shared__ __align__(16) _Float16 wt[32 * 32];

    const int wave = threadIdx.x >> 5;   // wave32: 8 waves / block
    const int lane = threadIdx.x & 31;
    const int half = lane >> 4;
    const int lr   = lane & 15;
    const int m0   = blockIdx.y * 32;
    const int pBlk = blockIdx.x * 128;

    v8f c0 = {}, c1 = {};

    for (int k0 = 0; k0 < Kpad; k0 += 32) {
        // ---- stage X tile: 128 rows x 64B ----
#ifdef USE_ASYNC_LDS
#pragma unroll
        for (int i = 0; i < 2; ++i) {
            int u = threadIdx.x + i * 256;
            int r = u >> 2, q = u & 3;
            const _Float16* src = X + (size_t)(pBlk + r) * Kpad + k0 + q * 8;
            __builtin_amdgcn_global_load_async_to_lds_b128(GAS1(src),
                                                           LAS3(xs + r * 32 + q * 8), 0, 0);
        }
        // ---- stage W tile: 32 rows x 64B ----
        if (threadIdx.x < 128) {
            int r = threadIdx.x >> 2, q = threadIdx.x & 3;
            const _Float16* src = Wf + (size_t)(m0 + r) * Kpad + k0 + q * 8;
            __builtin_amdgcn_global_load_async_to_lds_b128(GAS1(src),
                                                           LAS3(wt + r * 32 + q * 8), 0, 0);
        }
#else
#pragma unroll
        for (int i = 0; i < 2; ++i) {
            int u = threadIdx.x + i * 256;
            int r = u >> 2, q = u & 3;
            const int4* src = (const int4*)(X + (size_t)(pBlk + r) * Kpad + k0);
            ((int4*)(xs + r * 32))[q] = src[q];
        }
        if (threadIdx.x < 128) {
            int r = threadIdx.x >> 2, q = threadIdx.x & 3;
            const int4* src = (const int4*)(Wf + (size_t)(m0 + r) * Kpad + k0);
            ((int4*)(wt + r * 32))[q] = src[q];
        }
#endif
        if (k0 + 32 < Kpad) {
            // next K tile -> global_prefetch_b8
            __builtin_prefetch(X + (size_t)(pBlk + (threadIdx.x >> 1)) * Kpad + k0 + 32, 0, 1);
        }
#ifdef USE_ASYNC_LDS
        __builtin_amdgcn_s_wait_asynccnt(0);
#endif
        __syncthreads();

        // A tiles (16x32 f16, ISA 7.12.2): VGPR j 0..3 -> K=2j..+1 + 8*half,
        // VGPR 4..7 -> +16. Tile0 = wt rows 0..15, Tile1 = wt rows 16..31.
        v16h a0, a1, bmat;
#pragma unroll
        for (int j = 0; j < 4; ++j) {
            int kb0 = 2 * j + 8 * half;
            int kb1 = kb0 + 16;
            a0[2 * j]     = wt[lr * 32 + kb0];
            a0[2 * j + 1] = wt[lr * 32 + kb0 + 1];
            a0[8 + 2 * j]     = wt[lr * 32 + kb1];
            a0[8 + 2 * j + 1] = wt[lr * 32 + kb1 + 1];
            a1[2 * j]     = wt[(lr + 16) * 32 + kb0];
            a1[2 * j + 1] = wt[(lr + 16) * 32 + kb0 + 1];
            a1[8 + 2 * j]     = wt[(lr + 16) * 32 + kb1];
            a1[8 + 2 * j + 1] = wt[(lr + 16) * 32 + kb1 + 1];
        }
        // B (32x16 f16): same K order as A (elem 0..7 -> K=8*half+s, 8..15 -> +16), N=lr
        int prow = (wave * 16 + lr) * 32;
#pragma unroll
        for (int j = 0; j < 8; ++j) {
            int kk = 2 * (j & 3) + 8 * half + 16 * (j >> 2);
            bmat[2 * j]     = xs[prow + kk];
            bmat[2 * j + 1] = xs[prow + kk + 1];
        }
        c0 = __builtin_amdgcn_wmma_f32_16x16x32_f16(false, a0, false, bmat,
                                                    (short)0, c0, false, false);
        c1 = __builtin_amdgcn_wmma_f32_16x16x32_f16(false, a1, false, bmat,
                                                    (short)0, c1, false, false);
        __syncthreads();
    }

    int p = pBlk + wave * 16 + lr;
    if (p >= P) return;
#pragma unroll
    for (int vv = 0; vv < 8; ++vv) {
        int mA = m0 + vv + 8 * half;        // D layout: VGPR v -> M = v + 8*(lane>=16)
        int mB = mA + 16;
        if (mA < Mvalid) {
            float val = c0[vv] + bias[mA];
            if (relu) val = fmaxf(val, 0.f);
            if (out_f32) ((float*)Yv)[(size_t)p * Mstride + mA] = val;
            else ((_Float16*)Yv)[(size_t)p * Mstride + mA] = (_Float16)val;
        }
        if (mB < Mvalid) {
            float val = c1[vv] + bias[mB];
            if (relu) val = fmaxf(val, 0.f);
            if (out_f32) ((float*)Yv)[(size_t)p * Mstride + mB] = val;
            else ((_Float16*)Yv)[(size_t)p * Mstride + mB] = (_Float16)val;
        }
    }
}

// ---------------------------------------------------------------------------
// Small helper kernels
// ---------------------------------------------------------------------------
__global__ void build_x0(const float* __restrict__ pc, _Float16* __restrict__ X, int rows)
{
    int r = blockIdx.x * blockDim.x + threadIdx.x;
    if (r >= rows) return;
    const float* p = pc + (size_t)r * 3;
    _Float16* x = X + (size_t)r * 32;
    x[0] = (_Float16)p[0]; x[1] = (_Float16)p[1]; x[2] = (_Float16)p[2];
    for (int ch = 3; ch < 32; ++ch) x[ch] = (_Float16)0.f;
}

__global__ void maxpool_kernel(const _Float16* __restrict__ X, _Float16* __restrict__ Y,
                               int G, int NS, int C)
{
    int t = blockIdx.x * blockDim.x + threadIdx.x;
    if (t >= G * C) return;
    int g = t / C, ch = t % C;
    float mx = -3.0e38f;
    const _Float16* p = X + ((size_t)g * NS) * C + ch;
    for (int k = 0; k < NS; ++k) mx = fmaxf(mx, (float)p[(size_t)k * C]);
    Y[(size_t)g * C + ch] = (_Float16)mx;
}

__global__ void build_fcin(const _Float16* __restrict__ pool, int Cp,
                           const float* __restrict__ onehot,
                           _Float16* __restrict__ X, int Cpad, int Bq)
{
    int t = blockIdx.x * blockDim.x + threadIdx.x;
    if (t >= Bq * Cpad) return;
    int b = t / Cpad, ch = t % Cpad;
    float v;
    if (ch < Cp)           v = (float)pool[(size_t)b * Cp + ch];
    else if (ch < Cp + 3)  v = onehot[b * 3 + (ch - Cp)];
    else                   v = 0.f;
    X[t] = (_Float16)v;
}

__global__ void mean_kernel(const float* __restrict__ pc, float* __restrict__ meanv,
                            int Bq, int n)
{
    int t = blockIdx.x * blockDim.x + threadIdx.x;
    if (t >= Bq * 3) return;
    int d = t % 3, b = t / 3;
    float s = 0.f;
    for (int j = 0; j < n; ++j) s += pc[((size_t)b * n + j) * 3 + d];
    meanv[t] = s / (float)n;
}

__global__ void center_out(const float* __restrict__ delta, const float* __restrict__ meanv,
                           float* __restrict__ outc, int Bq)
{
    int t = blockIdx.x * blockDim.x + threadIdx.x;
    if (t < Bq * 3) outc[t] = delta[t] + meanv[t];
}

__global__ void recenter_kernel(const float* __restrict__ pc, const float* __restrict__ delta,
                                float* __restrict__ xyz, int Bq, int n)
{
    int t = blockIdx.x * blockDim.x + threadIdx.x;
    if (t >= Bq * n * 3) return;
    int d = t % 3, b = t / (n * 3);
    xyz[t] = pc[t] - delta[b * 3 + d];
}

// Farthest point sampling: one block per batch, sequential scan with LDS argmax.
__global__ void fps_kernel(const float* __restrict__ xyz, int n, int S, int* __restrict__ out)
{
    __shared__ float dists[2048];
    __shared__ float sval[256];
    __shared__ int   sidx[256];
    __shared__ int   sfar;
    __shared__ float cx, cy, cz;
    int b = blockIdx.x, t = threadIdx.x;
    for (int j = t; j < n; j += 256) dists[j] = 1e10f;
    if (t == 0) sfar = 0;
    __syncthreads();
    for (int it = 0; it < S; ++it) {
        if (t == 0) {
            out[b * S + it] = sfar;
            const float* c = xyz + ((size_t)b * n + sfar) * 3;
            cx = c[0]; cy = c[1]; cz = c[2];
        }
        __syncthreads();
        float bv = -1.f; int bi = 0;
        for (int j = t; j < n; j += 256) {
            const float* p = xyz + ((size_t)b * n + j) * 3;
            float dx = p[0] - cx, dy = p[1] - cy, dz = p[2] - cz;
            float nd = fminf(dists[j], dx * dx + dy * dy + dz * dz);
            dists[j] = nd;
            if (nd > bv) { bv = nd; bi = j; }
        }
        sval[t] = bv; sidx[t] = bi;
        __syncthreads();
        for (int s2 = 128; s2 > 0; s2 >>= 1) {
            if (t < s2) {
                if (sval[t + s2] > sval[t] ||
                    (sval[t + s2] == sval[t] && sidx[t + s2] < sidx[t])) {
                    sval[t] = sval[t + s2]; sidx[t] = sidx[t + s2];
                }
            }
            __syncthreads();
        }
        if (t == 0) sfar = sidx[0];
        __syncthreads();
    }
}

__global__ void gather_nxyz(const float* __restrict__ xyz, int n, const int* __restrict__ fidx,
                            float* __restrict__ nxyz, int S, int Bq)
{
    int t = blockIdx.x * blockDim.x + threadIdx.x;
    if (t >= Bq * S * 3) return;
    int d = t % 3, s = (t / 3) % S, b = t / (3 * S);
    nxyz[t] = xyz[((size_t)b * n + fidx[b * S + s]) * 3 + d];
}

// Ball query: first NS points within radius, pad with first hit (or 0 if none).
__global__ void ballq_kernel(const float* __restrict__ xyz, int n,
                             const float* __restrict__ nxyz, int S, int NS, float r2,
                             int* __restrict__ idx, int Bq)
{
    int t = blockIdx.x * blockDim.x + threadIdx.x;
    if (t >= Bq * S) return;
    int b = t / S;
    const float* c = nxyz + (size_t)t * 3;
    int* o = idx + (size_t)t * NS;
    int cnt = 0, first = 0; bool has = false;
    float cx = c[0], cy = c[1], cz = c[2];
    for (int j = 0; j < n && cnt < NS; ++j) {
        const float* p = xyz + ((size_t)b * n + j) * 3;
        float dx = p[0] - cx, dy = p[1] - cy, dz = p[2] - cz;
        if (dx * dx + dy * dy + dz * dz < r2) {
            if (!has) { first = j; has = true; }
            o[cnt++] = j;
        }
    }
    for (; cnt < NS; ++cnt) o[cnt] = first;
}

__global__ void build_group(const float* __restrict__ xyz, int n,
                            const float* __restrict__ nxyz,
                            const _Float16* __restrict__ feat, int Cf,
                            const int* __restrict__ idx, _Float16* __restrict__ X,
                            int S, int NS, int Cpad, int Bq)
{
    int r = blockIdx.x * blockDim.x + threadIdx.x;
    int total = Bq * S * NS;
    if (r >= total) return;
    int s = (r / NS) % S, b = r / (NS * S);
    int id = idx[r];
    const float* ps = xyz + ((size_t)b * n + id) * 3;
    const float* pcent = nxyz + ((size_t)b * S + s) * 3;
    _Float16* x = X + (size_t)r * Cpad;
    x[0] = (_Float16)(ps[0] - pcent[0]);
    x[1] = (_Float16)(ps[1] - pcent[1]);
    x[2] = (_Float16)(ps[2] - pcent[2]);
    if (feat) {
        const _Float16* f = feat + ((size_t)b * n + id) * Cf;
        for (int ch = 0; ch < Cf; ++ch) x[3 + ch] = f[ch];
    }
    for (int ch = 3 + Cf; ch < Cpad; ++ch) x[ch] = (_Float16)0.f;
}

__global__ void build_groupall(const float* __restrict__ nxyz,
                               const _Float16* __restrict__ feat, int Cf,
                               _Float16* __restrict__ X, int S, int Cpad, int Bq)
{
    int r = blockIdx.x * blockDim.x + threadIdx.x;
    if (r >= Bq * S) return;
    const float* p = nxyz + (size_t)r * 3;
    const _Float16* f = feat + (size_t)r * Cf;
    _Float16* x = X + (size_t)r * Cpad;
    x[0] = (_Float16)p[0]; x[1] = (_Float16)p[1]; x[2] = (_Float16)p[2];
    for (int ch = 0; ch < Cf; ++ch) x[3 + ch] = f[ch];
    for (int ch = 3 + Cf; ch < Cpad; ++ch) x[ch] = (_Float16)0.f;
}

// ---------------------------------------------------------------------------
// Host orchestration
// ---------------------------------------------------------------------------
extern "C" void kernel_launch(void* const* d_in, const int* in_sizes, int n_in,
                              void* d_out, int out_size, void* d_ws, size_t ws_size,
                              hipStream_t stream)
{
    (void)in_sizes; (void)n_in; (void)out_size; (void)ws_size;
#define FP(k) ((const float*)d_in[(k)])
    const float* pc     = FP(0);   // [128, 2048, 3]
    const float* onehot = FP(1);   // [128, 3]

    char* base = (char*)d_ws;
    size_t off = 0;
    auto alloc = [&](size_t bytes) -> void* {
        off = (off + 255) & ~(size_t)255;
        void* p = base + off;
        off += bytes;
        return p;
    };

    const int BN = BB * NN;
    float* meanv = (float*)alloc(BB * 3 * 4);
    float* delta = (float*)alloc(BB * 3 * 4);
    float* xyzr  = (float*)alloc((size_t)BN * 3 * 4);
    int*   fidx1 = (int*)alloc(BB * S1 * 4);
    float* nxyz1 = (float*)alloc((size_t)BB * S1 * 3 * 4);
    int*   bidx1 = (int*)alloc((size_t)BB * S1 * NS1 * 4);
    _Float16* l1f = (_Float16*)alloc((size_t)BB * S1 * 128 * 2);
    int*   fidx2 = (int*)alloc(BB * S2 * 4);
    float* nxyz2 = (float*)alloc((size_t)BB * S2 * 3 * 4);
    int*   bidx2 = (int*)alloc((size_t)BB * S2 * NS2 * 4);
    _Float16* l2f   = (_Float16*)alloc((size_t)BB * S2 * 256 * 2);
    _Float16* pool3 = (_Float16*)alloc(BB * 512 * 2);
    _Float16* pstn  = (_Float16*)alloc(BB * 256 * 2);
    _Float16* stnin = (_Float16*)alloc(BB * 288 * 2);
    _Float16* stnh1 = (_Float16*)alloc(BB * 256 * 2);
    _Float16* stnh2 = (_Float16*)alloc(BB * 128 * 2);
    _Float16* fcin  = (_Float16*)alloc(BB * 544 * 2);
    _Float16* fch1  = (_Float16*)alloc(BB * 512 * 2);
    _Float16* fch2  = (_Float16*)alloc(BB * 256 * 2);

    struct FW { _Float16* w; float* b; };
    auto fwalloc = [&](int Mpad, int Kpad) -> FW {
        FW f;
        f.w = (_Float16*)alloc((size_t)Mpad * Kpad * 2);
        f.b = (float*)alloc(Mpad * 4);
        return f;
    };
    // All Mpad multiples of 32 (wave computes 32 Cout rows)
    FW stn_c1 = fwalloc(128, 32),  stn_c2 = fwalloc(128, 128), stn_c3 = fwalloc(256, 128);
    FW stn_f1 = fwalloc(256, 288), stn_f2 = fwalloc(128, 256), stn_f3 = fwalloc(32, 128);
    FW s1a = fwalloc(64, 32),   s1b = fwalloc(64, 64),   s1c = fwalloc(128, 64);
    FW s2a = fwalloc(128, 160), s2b = fwalloc(128, 128), s2c = fwalloc(256, 128);
    FW s3a = fwalloc(256, 288), s3b = fwalloc(256, 256), s3c = fwalloc(512, 256);
    FW ff1 = fwalloc(512, 544), ff2 = fwalloc(256, 512), ff3 = fwalloc(64, 256);
    size_t arena = off;

    auto fold = [&](const float* W, const float* bc, const float* g, const float* bb_,
                    const float* m, const float* v, FW fw, int M, int K, int Mpad, int Kpad) {
        int tot = Mpad * Kpad;
        fold_bn_kernel<<<(tot + 255) / 256, 256, 0, stream>>>(W, bc, g, bb_, m, v,
                                                              fw.w, fw.b, M, K, Mpad, Kpad);
    };
    auto gemm = [&](FW fw, const _Float16* X, void* Y, int P, int Kpad, int Mpad,
                    int Mvalid, int Mstride, int relu, int f32out) {
        dim3 g(P / 128, Mpad / 32);
        gemm_wmma<<<g, 256, 0, stream>>>(fw.w, fw.b, X, Y, P, Kpad, Mvalid, Mstride, relu, f32out);
    };

    // ---- fold all layers (param leaf indices: JAX sorted-key order) ----
    fold(FP(83), FP(82), FP(71), FP(70), FP(72), FP(73), stn_c1, 128, 3, 128, 32);
    fold(FP(85), FP(84), FP(75), FP(74), FP(76), FP(77), stn_c2, 128, 128, 128, 128);
    fold(FP(87), FP(86), FP(79), FP(78), FP(80), FP(81), stn_c3, 256, 128, 256, 128);
    fold(FP(89), FP(88), FP(95), FP(94), FP(96), FP(97), stn_f1, 256, 259, 256, 288);
    fold(FP(91), FP(90), FP(99), FP(98), FP(100), FP(101), stn_f2, 128, 256, 128, 256);
    fold(FP(93), FP(92), nullptr, nullptr, nullptr, nullptr, stn_f3, 3, 128, 32, 128);
    fold(FP(21), FP(20), FP(17), FP(16), FP(18), FP(19), s1a, 64, 3, 64, 32);
    fold(FP(27), FP(26), FP(23), FP(22), FP(24), FP(25), s1b, 64, 64, 64, 64);
    fold(FP(33), FP(32), FP(29), FP(28), FP(30), FP(31), s1c, 128, 64, 128, 64);
    fold(FP(39), FP(38), FP(35), FP(34), FP(36), FP(37), s2a, 128, 131, 128, 160);
    fold(FP(45), FP(44), FP(41), FP(40), FP(42), FP(43), s2b, 128, 128, 128, 128);
    fold(FP(51), FP(50), FP(47), FP(46), FP(48), FP(49), s2c, 256, 128, 256, 128);
    fold(FP(57), FP(56), FP(53), FP(52), FP(54), FP(55), s3a, 256, 259, 256, 288);
    fold(FP(63), FP(62), FP(59), FP(58), FP(60), FP(61), s3b, 256, 256, 256, 256);
    fold(FP(69), FP(68), FP(65), FP(64), FP(66), FP(67), s3c, 512, 256, 512, 256);
    fold(FP(11), FP(10), FP(3), FP(2), FP(4), FP(5), ff1, 512, 515, 512, 544);
    fold(FP(13), FP(12), FP(7), FP(6), FP(8), FP(9), ff2, 256, 512, 256, 512);
    fold(FP(15), FP(14), nullptr, nullptr, nullptr, nullptr, ff3, 59, 256, 64, 256);

    // ---- Phase A: STN ----
    off = arena;
    _Float16* X0 = (_Float16*)alloc((size_t)BN * 32 * 2);
    _Float16* a1 = (_Float16*)alloc((size_t)BN * 128 * 2);
    _Float16* a2 = (_Float16*)alloc((size_t)BN * 128 * 2);
    _Float16* a3 = (_Float16*)alloc((size_t)BN * 256 * 2);
    build_x0<<<(BN + 255) / 256, 256, 0, stream>>>(pc, X0, BN);
    gemm(stn_c1, X0, a1, BN, 32, 128, 128, 128, 1, 0);
    gemm(stn_c2, a1, a2, BN, 128, 128, 128, 128, 1, 0);
    gemm(stn_c3, a2, a3, BN, 128, 256, 256, 256, 1, 0);
    maxpool_kernel<<<(BB * 256 + 255) / 256, 256, 0, stream>>>(a3, pstn, BB, NN, 256);
    build_fcin<<<(BB * 288 + 255) / 256, 256, 0, stream>>>(pstn, 256, onehot, stnin, 288, BB);
    gemm(stn_f1, stnin, stnh1, BB, 288, 256, 256, 256, 1, 0);
    gemm(stn_f2, stnh1, stnh2, BB, 256, 128, 128, 128, 1, 0);
    gemm(stn_f3, stnh2, delta, BB, 128, 32, 3, 3, 0, 1);
    mean_kernel<<<(BB * 3 + 255) / 256, 256, 0, stream>>>(pc, meanv, BB, NN);
    center_out<<<(BB * 3 + 255) / 256, 256, 0, stream>>>(delta, meanv,
                                                         (float*)d_out + BB * 59, BB);
    recenter_kernel<<<(BN * 3 + 255) / 256, 256, 0, stream>>>(pc, delta, xyzr, BB, NN);

    // ---- Phase B: SA1 (npoint=128, r=0.2, ns=64, MLP 3->64->64->128) ----
    off = arena;
    const int P1 = BB * S1 * NS1;  // 1,048,576
    _Float16* Xg1 = (_Float16*)alloc((size_t)P1 * 32 * 2);
    _Float16* h1a = (_Float16*)alloc((size_t)P1 * 64 * 2);
    _Float16* h1b = (_Float16*)alloc((size_t)P1 * 64 * 2);
    _Float16* h1c = (_Float16*)alloc((size_t)P1 * 128 * 2);
    fps_kernel<<<BB, 256, 0, stream>>>(xyzr, NN, S1, fidx1);
    gather_nxyz<<<(BB * S1 * 3 + 255) / 256, 256, 0, stream>>>(xyzr, NN, fidx1, nxyz1, S1, BB);
    ballq_kernel<<<(BB * S1 + 255) / 256, 256, 0, stream>>>(xyzr, NN, nxyz1, S1, NS1,
                                                            0.2f * 0.2f, bidx1, BB);
    build_group<<<(P1 + 255) / 256, 256, 0, stream>>>(xyzr, NN, nxyz1, nullptr, 0,
                                                      bidx1, Xg1, S1, NS1, 32, BB);
    gemm(s1a, Xg1, h1a, P1, 32, 64, 64, 64, 1, 0);
    gemm(s1b, h1a, h1b, P1, 64, 64, 64, 64, 1, 0);
    gemm(s1c, h1b, h1c, P1, 64, 128, 128, 128, 1, 0);
    maxpool_kernel<<<(BB * S1 * 128 + 255) / 256, 256, 0, stream>>>(h1c, l1f, BB * S1, NS1, 128);

    // ---- Phase C: SA2 (npoint=32, r=0.4, ns=64, MLP 131->128->128->256) ----
    off = arena;
    const int P2 = BB * S2 * NS2;  // 262,144
    _Float16* Xg2 = (_Float16*)alloc((size_t)P2 * 160 * 2);
    _Float16* h2a = (_Float16*)alloc((size_t)P2 * 128 * 2);
    _Float16* h2b = (_Float16*)alloc((size_t)P2 * 128 * 2);
    _Float16* h2c = (_Float16*)alloc((size_t)P2 * 256 * 2);
    fps_kernel<<<BB, 256, 0, stream>>>(nxyz1, S1, S2, fidx2);
    gather_nxyz<<<(BB * S2 * 3 + 255) / 256, 256, 0, stream>>>(nxyz1, S1, fidx2, nxyz2, S2, BB);
    ballq_kernel<<<(BB * S2 + 255) / 256, 256, 0, stream>>>(nxyz1, S1, nxyz2, S2, NS2,
                                                            0.4f * 0.4f, bidx2, BB);
    build_group<<<(P2 + 255) / 256, 256, 0, stream>>>(nxyz1, S1, nxyz2, l1f, 128,
                                                      bidx2, Xg2, S2, NS2, 160, BB);
    gemm(s2a, Xg2, h2a, P2, 160, 128, 128, 128, 1, 0);
    gemm(s2b, h2a, h2b, P2, 128, 128, 128, 128, 1, 0);
    gemm(s2c, h2b, h2c, P2, 128, 256, 256, 256, 1, 0);
    maxpool_kernel<<<(BB * S2 * 256 + 255) / 256, 256, 0, stream>>>(h2c, l2f, BB * S2, NS2, 256);

    // ---- Phase D: SA3 group-all (MLP 259->256->256->512) + head FCs ----
    off = arena;
    const int P3 = BB * S2;  // 4096
    _Float16* Xg3 = (_Float16*)alloc((size_t)P3 * 288 * 2);
    _Float16* g1 = (_Float16*)alloc((size_t)P3 * 256 * 2);
    _Float16* g2 = (_Float16*)alloc((size_t)P3 * 256 * 2);
    _Float16* g3 = (_Float16*)alloc((size_t)P3 * 512 * 2);
    build_groupall<<<(P3 + 255) / 256, 256, 0, stream>>>(nxyz2, l2f, 256, Xg3, S2, 288, BB);
    gemm(s3a, Xg3, g1, P3, 288, 256, 256, 256, 1, 0);
    gemm(s3b, g1, g2, P3, 256, 256, 256, 256, 1, 0);
    gemm(s3c, g2, g3, P3, 256, 512, 512, 512, 1, 0);
    maxpool_kernel<<<(BB * 512 + 255) / 256, 256, 0, stream>>>(g3, pool3, BB, S2, 512);
    build_fcin<<<(BB * 544 + 255) / 256, 256, 0, stream>>>(pool3, 512, onehot, fcin, 544, BB);
    gemm(ff1, fcin, fch1, BB, 544, 512, 512, 512, 1, 0);
    gemm(ff2, fch1, fch2, BB, 512, 256, 256, 256, 1, 0);
    gemm(ff3, fch2, d_out, BB, 256, 64, 59, 59, 0, 1);  // box_pred f32 -> d_out[0..128*59)
#undef FP
}
